// WorldModel_16088947491309
// MI455X (gfx1250) — compile-verified, matrix operand order checked by reading, not run on hardware
//
#include <hip/hip_runtime.h>
#include <hip/hip_bf16.h>
#include <math.h>

// ---------------- problem constants ----------------
#define TPF     36
#define KF      4
#define DM      1024
#define NH      16
#define HD      64
#define NLAYER  8
#define VOC     4096
#define BB      64
#define SEQ     184          // KF*(TPF+1)+TPF
#define CTX_END 148          // KF*(TPF+1)

static_assert(SEQ == KF * (TPF + 1) + TPF, "seq");

typedef _Float16 half_t;
typedef __attribute__((ext_vector_type(16))) _Float16 v16h;
typedef __attribute__((ext_vector_type(8)))  float    v8f;
typedef unsigned int u32x4 __attribute__((ext_vector_type(4)));
typedef int i32x8 __attribute__((ext_vector_type(8)));
typedef int i32x4 __attribute__((ext_vector_type(4)));

// Tensor Data Mover availability (guarded: compiles clean either way)
#if defined(__has_builtin)
#if __has_builtin(__builtin_amdgcn_tensor_load_to_lds) && __has_builtin(__builtin_amdgcn_s_wait_tensorcnt)
#define USE_TDM 1
#endif
#endif

// ---------------- helpers ----------------
__device__ __forceinline__ v16h load_frag(const half_t* p) {
  // 16-bit WMMA A/B fragment: two contiguous 8-half chunks (K base, K base+16)
  union { uint4 u[2]; v16h v; } f;
  f.u[0] = *(const uint4*)(p);
  f.u[1] = *(const uint4*)(p + 16);
  return f.v;
}

__device__ __forceinline__ float gelu_exact(float x) {
  return 0.5f * x * (1.0f + erff(x * 0.70710678118654752f));
}

__device__ __forceinline__ int block_of(int p) {
  return p < CTX_END ? 2 * (p / (TPF + 1)) + ((p % (TPF + 1)) == TPF) : 2 * KF;
}
__device__ __forceinline__ bool is_masked(int q, int k) {
  int bq = block_of(q), bk = block_of(k);
  bool m = bq < bk;
  if (q >= CTX_END && k >= CTX_END && k > q) m = true;
  return m;
}

// ---------------- f32 -> f16 conversion (weights staging) ----------------
__global__ __launch_bounds__(256) void cvt_kernel(const float* __restrict__ in,
                                                  half_t* __restrict__ out, int n4) {
  int i = blockIdx.x * 256 + threadIdx.x;
  if (i < n4) {
    float4 v = ((const float4*)in)[i];
    union { half_t h[4]; uint2 u; } o;
    o.h[0] = (half_t)v.x; o.h[1] = (half_t)v.y; o.h[2] = (half_t)v.z; o.h[3] = (half_t)v.w;
    ((uint2*)out)[i] = o.u;
  }
}

// ---------------- embedding (residual stream stays f32) ----------------
__global__ __launch_bounds__(256) void embed_kernel(
    float* __restrict__ x, const int* __restrict__ ft, const int* __restrict__ act,
    const int* __restrict__ lvl, const float* __restrict__ tok,
    const float* __restrict__ ae, const float* __restrict__ le) {
  int bs = blockIdx.x;
  int b = bs / SEQ, pos = bs % SEQ;
  const float* src;
  if (pos < CTX_END) {
    int i = pos / (TPF + 1), j = pos % (TPF + 1);
    if (j == TPF) src = ae + (size_t)act[b * KF + i] * DM;
    else          src = tok + (size_t)ft[(b * (KF + 1) + i) * TPF + j] * DM;
  } else {
    src = tok + (size_t)ft[(b * (KF + 1) + KF) * TPF + (pos - CTX_END)] * DM;
  }
  const float* lrow = le + (size_t)lvl[b] * DM;
  float4 s4 = ((const float4*)src)[threadIdx.x];
  float4 l4 = ((const float4*)lrow)[threadIdx.x];
  float4 o; o.x = s4.x + l4.x; o.y = s4.y + l4.y; o.z = s4.z + l4.z; o.w = s4.w + l4.w;
  ((float4*)(x + (size_t)bs * DM))[threadIdx.x] = o;
}

// ---------------- layernorm: f32 in, f16 out ----------------
__global__ __launch_bounds__(256) void ln_kernel(
    const float* __restrict__ x, const float* __restrict__ g,
    const float* __restrict__ b, half_t* __restrict__ out) {
  __shared__ float s1[256], s2[256];
  int row = blockIdx.x, t = threadIdx.x;
  const float* xr = x + (size_t)row * DM;
  float4 xv = ((const float4*)xr)[t];
  float s  = xv.x + xv.y + xv.z + xv.w;
  float ss = xv.x * xv.x + xv.y * xv.y + xv.z * xv.z + xv.w * xv.w;
  s1[t] = s; s2[t] = ss; __syncthreads();
  for (int o = 128; o > 0; o >>= 1) {
    if (t < o) { s1[t] += s1[t + o]; s2[t] += s2[t + o]; }
    __syncthreads();
  }
  float mean = s1[0] * (1.0f / DM);
  float var  = s2[0] * (1.0f / DM) - mean * mean;
  float rstd = rsqrtf(var + 1e-5f);
  float4 gv = ((const float4*)g)[t], bv = ((const float4*)b)[t];
  union { half_t h[4]; uint2 u; } o4;
  o4.h[0] = (half_t)((xv.x - mean) * rstd * gv.x + bv.x);
  o4.h[1] = (half_t)((xv.y - mean) * rstd * gv.y + bv.y);
  o4.h[2] = (half_t)((xv.z - mean) * rstd * gv.z + bv.z);
  o4.h[3] = (half_t)((xv.w - mean) * rstd * gv.w + bv.w);
  ((uint2*)(out + (size_t)row * DM))[t] = o4.u;
}

// ---------------- WMMA GEMM: C[M,N] = act(A @ W^T + bias) (+C) ----------------
// A: f16 [M,Kd] row-major, W: f16 [N,Kd] row-major. 128x256 block tile,
// 8 waves of 64x64, BK=64, LDS tiles padded to 72-half stride.
#define BM 128
#define BN 256
#define BK 64
#define LDT 72
#define BUF_HALVES ((BM + BN) * LDT)        // one [A|B] buffer, in halves
#define BUF_BYTES  (BUF_HALVES * 2)
#define GSMEM      (2 * BUF_BYTES)          // double buffered
#define FLAG_GELU  1
#define FLAG_RESID 2
#define FLAG_OUT16 4

#ifdef USE_TDM
// Issue one 2-D TDM tile load: tile_d1 rows x tile_d0 halves, row stride
// stride0 (halves) in memory, padded in LDS every 128B by 16B (-> 72-half rows).
__device__ __forceinline__ void tdm_load_2d(const half_t* gptr, unsigned lds_addr,
                                            int tile_d0, int tile_d1,
                                            int tensor_d0, int tensor_d1, int stride0) {
  unsigned long long ga = (unsigned long long)(size_t)gptr;
  u32x4 g0 = {0u, 0u, 0u, 0u};
  g0[0] = 1u;                                         // count=1 (valid), user mode
  g0[1] = lds_addr;                                   // LDS byte address
  g0[2] = (unsigned)(ga & 0xffffffffu);               // global_addr[31:0]
  g0[3] = (unsigned)((ga >> 32) & 0x1ffffffu) | (2u << 30);  // addr[56:32] | type=2
  unsigned w0 = (1u << 16)    // data_size = 2 bytes
              | (1u << 20)    // pad_enable
              | (4u << 22)    // pad_interval: 32 DWORDs (128B)
              | (3u << 25);   // pad_amount: 4 DWORDs (16B)
  i32x8 g1 = {0, 0, 0, 0, 0, 0, 0, 0};
  g1[0] = (int)w0;
  g1[1] = (int)((unsigned)tensor_d0 << 16);
  g1[2] = (int)(((unsigned)tensor_d0 >> 16) | ((unsigned)tensor_d1 << 16));
  g1[3] = (int)(((unsigned)tensor_d1 >> 16) | ((unsigned)tile_d0 << 16));
  g1[4] = (int)((unsigned)tile_d1 & 0xffffu);         // tile_dim1 (tile_dim2 = 0)
  g1[5] = (int)(unsigned)stride0;                     // tensor_dim0_stride[31:0]
  g1[6] = 0;
  g1[7] = 0;
  i32x4 g2 = {0, 0, 0, 0};
  i32x4 g3 = {0, 0, 0, 0};
#if __clang_major__ >= 23
  i32x8 g4 = {0, 0, 0, 0, 0, 0, 0, 0};
  __builtin_amdgcn_tensor_load_to_lds(g0, g1, g2, g3, g4, 0);
#else
  __builtin_amdgcn_tensor_load_to_lds(g0, g1, g2, g3, 0);
#endif
}
__device__ __forceinline__ unsigned lds_off(const void* p) {
  return (unsigned)(unsigned long long)(size_t)p;     // low 32 bits = LDS byte offset
}
#endif

__global__ __launch_bounds__(256) void gemm_f16(
    const half_t* __restrict__ A, const half_t* __restrict__ W,
    const float* __restrict__ bias, void* __restrict__ Cv,
    int M, int N, int Kd, int flags) {
  extern __shared__ char dynsmem[];
  half_t* sbase = (half_t*)dynsmem;                   // [A|B] x 2 buffers

  const int t = threadIdx.x;
  const int lane = t & 31, wave = t >> 5;
  const int wr = wave >> 2, wc = wave & 3;            // 2x4 waves of 64x64
  const int hl = lane >> 4, ll = lane & 15;
  const int m0 = blockIdx.x * BM, n0 = blockIdx.y * BN;
  const int ksteps = Kd / BK;

  // per-wave fragment base offsets (halves, within one buffer)
  const int aoff = (wr * 64 + ll) * LDT + hl * 8;
  const int boff = BM * LDT + (wc * 64 + ll) * LDT + hl * 8;

  v8f vzero = {};
  v8f acc[4][4];
#pragma unroll
  for (int mt = 0; mt < 4; mt++)
#pragma unroll
    for (int nt = 0; nt < 4; nt++) acc[mt][nt] = vzero;

  auto compute = [&](int bufoff) {
#pragma unroll
    for (int hk = 0; hk < 2; hk++) {
      const int ko = hk * 32;
      v16h af[4], bf[4];
#pragma unroll
      for (int mt = 0; mt < 4; mt++)
        af[mt] = load_frag(sbase + bufoff + aoff + mt * 16 * LDT + ko);
#pragma unroll
      for (int nt = 0; nt < 4; nt++)
        bf[nt] = load_frag(sbase + bufoff + boff + nt * 16 * LDT + ko);
#pragma unroll
      for (int mt = 0; mt < 4; mt++)
#pragma unroll
        for (int nt = 0; nt < 4; nt++)
          acc[mt][nt] = __builtin_amdgcn_wmma_f32_16x16x32_f16(
              false, af[mt], false, bf[nt], (short)0, acc[mt][nt], false, false);
    }
  };

#ifdef USE_TDM
  const half_t* Agt = A + (size_t)m0 * Kd;
  const half_t* Wgt = W + (size_t)n0 * Kd;
  const unsigned lbase = lds_off(sbase);
  if (wave == 0) {
    tdm_load_2d(Agt, lbase, BK, BM, Kd, M, Kd);
    tdm_load_2d(Wgt, lbase + BM * LDT * 2, BK, BN, Kd, N, Kd);
  }
  for (int ks = 0; ks < ksteps; ks++) {
    if (wave == 0) {
      if (ks + 1 < ksteps) {
        unsigned lb = lbase + (unsigned)((ks + 1) & 1) * BUF_BYTES;
        tdm_load_2d(Agt + (size_t)(ks + 1) * BK, lb, BK, BM, Kd, M, Kd);
        tdm_load_2d(Wgt + (size_t)(ks + 1) * BK, lb + BM * LDT * 2, BK, BN, Kd, N, Kd);
        __builtin_amdgcn_s_wait_tensorcnt(2);   // oldest pair (tile ks) complete
      } else {
        __builtin_amdgcn_s_wait_tensorcnt(0);
      }
    }
    __syncthreads();
    compute((ks & 1) * BUF_HALVES);
    __syncthreads();   // all reads done before TDM reuses this buffer
  }
#else
  // fallback: manual f16 tile staging, single buffered
  const int arow = t >> 1, acol = (t & 1) * 32;       // 32 halves of A per thread
  const half_t* Ag = A + (size_t)(m0 + arow) * Kd + acol;
  const half_t* Wg = W + (size_t)(n0 + t) * Kd;       // 64 halves of W per thread
  half_t* Asw = sbase + arow * LDT + acol;
  half_t* Bsw = sbase + BM * LDT + t * LDT;
  for (int ks = 0; ks < ksteps; ks++) {
    uint4 a0 = ((const uint4*)Ag)[0], a1 = ((const uint4*)Ag)[1];
    uint4 a2 = ((const uint4*)Ag)[2], a3 = ((const uint4*)Ag)[3];
    uint4 b0 = ((const uint4*)Wg)[0], b1 = ((const uint4*)Wg)[1];
    uint4 b2 = ((const uint4*)Wg)[2], b3 = ((const uint4*)Wg)[3];
    uint4 b4 = ((const uint4*)Wg)[4], b5 = ((const uint4*)Wg)[5];
    uint4 b6 = ((const uint4*)Wg)[6], b7 = ((const uint4*)Wg)[7];
    Ag += BK; Wg += BK;
    __builtin_prefetch(Ag);
    __builtin_prefetch(Wg);
    __syncthreads();
    ((uint4*)Asw)[0] = a0; ((uint4*)Asw)[1] = a1;
    ((uint4*)Asw)[2] = a2; ((uint4*)Asw)[3] = a3;
    ((uint4*)Bsw)[0] = b0; ((uint4*)Bsw)[1] = b1;
    ((uint4*)Bsw)[2] = b2; ((uint4*)Bsw)[3] = b3;
    ((uint4*)Bsw)[4] = b4; ((uint4*)Bsw)[5] = b5;
    ((uint4*)Bsw)[6] = b6; ((uint4*)Bsw)[7] = b7;
    __syncthreads();
    compute(0);
  }
#endif

  // epilogue: D element (m,n): m = r + 8*(lane/16), n = lane%16
  float*  Cf = (float*)Cv;
  half_t* Ch = (half_t*)Cv;
#pragma unroll
  for (int mt = 0; mt < 4; mt++) {
#pragma unroll
    for (int nt = 0; nt < 4; nt++) {
      int rbase = m0 + wr * 64 + mt * 16 + hl * 8;
      int cbase = n0 + wc * 64 + nt * 16 + ll;
      float bv = bias[cbase];
#pragma unroll
      for (int r = 0; r < 8; r++) {
        float v = acc[mt][nt][r] + bv;
        if (flags & FLAG_GELU) v = gelu_exact(v);
        size_t idx = (size_t)(rbase + r) * N + cbase;
        if (flags & FLAG_OUT16) {
          Ch[idx] = (half_t)v;
        } else {
          if (flags & FLAG_RESID) v += Cf[idx];
          Cf[idx] = v;
        }
      }
    }
  }
}

// ---------------- rope: f16 qkv -> rotated f16 Q/K + V, [b,h,s,d] ----------------
__global__ __launch_bounds__(256) void rope_kernel(
    const half_t* __restrict__ qkv, const float* __restrict__ rc,
    const float* __restrict__ rs, half_t* __restrict__ Qh,
    half_t* __restrict__ Kh, half_t* __restrict__ Vh) {
  int idx = blockIdx.x * 256 + threadIdx.x;       // over B*S*NH*HD
  int d = idx & 63;
  int h = (idx >> 6) & 15;
  int bs = idx >> 10;
  int b = bs / SEQ, s = bs % SEQ;
  const half_t* row = qkv + (size_t)bs * (3 * DM);
  size_t obase = ((size_t)(b * NH + h) * SEQ + s) * HD;
  Vh[obase + d] = row[2 * DM + h * HD + d];
  if (d < 32) {
    float c = rc[s * 32 + d], sn = rs[s * 32 + d];
    float q1 = (float)row[h * HD + d], q2 = (float)row[h * HD + d + 32];
    Qh[obase + d]      = (half_t)(q1 * c - q2 * sn);
    Qh[obase + d + 32] = (half_t)(q1 * sn + q2 * c);
    float k1 = (float)row[DM + h * HD + d], k2 = (float)row[DM + h * HD + d + 32];
    Kh[obase + d]      = (half_t)(k1 * c - k2 * sn);
    Kh[obase + d + 32] = (half_t)(k1 * sn + k2 * c);
  }
}

// ---------------- attention (WMMA QK^T, softmax, WMMA PV) ----------------
#define SP   192            // padded seq (12 x 16)
#define KLDA 72             // K LDS stride (halves)
#define VLDA 200            // transposed-V LDS stride (halves)
#define SLDA 196            // scores LDS stride (floats)
#define ATTN_SMEM ((SP*KLDA + HD*VLDA) * 2 + 6 * 16 * SLDA * 4)

__global__ __launch_bounds__(192) void attn_kernel(
    const half_t* __restrict__ Qh, const half_t* __restrict__ Kh,
    const half_t* __restrict__ Vh, half_t* __restrict__ outp) {
  extern __shared__ char dynsmem[];
  half_t* Ks = (half_t*)dynsmem;                      // [SP][KLDA] key x dim
  half_t* Vt = Ks + SP * KLDA;                        // [HD][VLDA] dim x key
  float*  Sc = (float*)(Vt + HD * VLDA);              // [6*16][SLDA]

  const int bh = blockIdx.x >> 1;
  const int half_id = blockIdx.x & 1;
  const int b = bh >> 4, h = bh & 15;
  const int t = threadIdx.x, lane = t & 31, wave = t >> 5;   // wave 0..5
  const int hl = lane >> 4, ll = lane & 15;

  const half_t* Kg = Kh + (size_t)bh * SEQ * HD;
  const half_t* Vg = Vh + (size_t)bh * SEQ * HD;

  {
    int r = t;
    int rc = r < SEQ ? r : SEQ - 1;
    bool valid = r < SEQ;
    uint4 z; z.x = z.y = z.z = z.w = 0u;
    const uint4* gk = (const uint4*)(Kg + (size_t)rc * HD);
    uint4* lk = (uint4*)(Ks + r * KLDA);
#pragma unroll
    for (int i = 0; i < 8; i++) lk[i] = valid ? gk[i] : z;
    union { uint4 u[8]; half_t hv[64]; } vb;
    const uint4* gv = (const uint4*)(Vg + (size_t)rc * HD);
#pragma unroll
    for (int i = 0; i < 8; i++) vb.u[i] = valid ? gv[i] : z;
#pragma unroll
    for (int d0 = 0; d0 < 64; d0++) Vt[d0 * VLDA + r] = vb.hv[d0];
  }
  __syncthreads();

  const int qt = half_id * 6 + wave;
  const int m0 = qt * 16;
  int qrow = m0 + ll; if (qrow >= SEQ) qrow = SEQ - 1;
  const half_t* qp = Qh + (size_t)bh * SEQ * HD + (size_t)qrow * HD;
  v16h qa0 = load_frag(qp + hl * 8);
  v16h qa1 = load_frag(qp + 32 + hl * 8);

  float* myS = Sc + wave * 16 * SLDA;
  const float scale = 0.125f;               // HD^-0.5

  for (int kt = 0; kt < 12; kt++) {
    v16h kb0 = load_frag(Ks + (kt * 16 + ll) * KLDA + hl * 8);
    v16h kb1 = load_frag(Ks + (kt * 16 + ll) * KLDA + 32 + hl * 8);
    v8f c = {};
    c = __builtin_amdgcn_wmma_f32_16x16x32_f16(false, qa0, false, kb0, (short)0, c, false, false);
    c = __builtin_amdgcn_wmma_f32_16x16x32_f16(false, qa1, false, kb1, (short)0, c, false, false);
    int kcol = kt * 16 + ll;
#pragma unroll
    for (int r = 0; r < 8; r++) {
      int ql = hl * 8 + r;
      int q = m0 + ql; if (q >= SEQ) q = SEQ - 1;
      float v = c[r] * scale;
      if (kcol >= SEQ || is_masked(q, kcol)) v = -__builtin_inff();
      myS[ql * SLDA + kcol] = v;
    }
  }

  if (lane < 16) {
    float* row = myS + lane * SLDA;
    float mx = -__builtin_inff();
    for (int k2 = 0; k2 < SEQ; k2++) mx = fmaxf(mx, row[k2]);
    float sum = 0.0f;
    for (int k2 = 0; k2 < SEQ; k2++) { float e = expf(row[k2] - mx); row[k2] = e; sum += e; }
    float inv = 1.0f / sum;
    for (int k2 = 0; k2 < SEQ; k2++) row[k2] *= inv;
    for (int k2 = SEQ; k2 < SP; k2++) row[k2] = 0.0f;
  }

  v8f oacc[4];
  v8f vz = {};
#pragma unroll
  for (int nt = 0; nt < 4; nt++) oacc[nt] = vz;
  for (int kt2 = 0; kt2 < 6; kt2++) {
    union { half_t hh[16]; v16h v; } pu;
    const float* srow = myS + ll * SLDA + kt2 * 32 + hl * 8;
#pragma unroll
    for (int j = 0; j < 8; j++) {
      pu.hh[j]     = (half_t)srow[j];
      pu.hh[8 + j] = (half_t)srow[16 + j];
    }
    v16h pa = pu.v;
#pragma unroll
    for (int nt = 0; nt < 4; nt++) {
      v16h vbf = load_frag(Vt + (nt * 16 + ll) * VLDA + kt2 * 32 + hl * 8);
      oacc[nt] = __builtin_amdgcn_wmma_f32_16x16x32_f16(false, pa, false, vbf, (short)0, oacc[nt], false, false);
    }
  }

#pragma unroll
  for (int nt = 0; nt < 4; nt++) {
#pragma unroll
    for (int r = 0; r < 8; r++) {
      int q = m0 + hl * 8 + r;
      if (q < SEQ) {
        int d = nt * 16 + ll;
        outp[((size_t)(b * SEQ + q)) * DM + h * HD + d] = (half_t)oacc[nt][r];
      }
    }
  }
}

// ---------------- gather predict rows (f16) / death head ----------------
__global__ __launch_bounds__(256) void gather_kernel(const half_t* __restrict__ hin,
                                                     half_t* __restrict__ outp) {
  int row = blockIdx.x;                 // 0..B*TPF-1
  int b = row / TPF, j = row % TPF;
  const uint2* src = (const uint2*)(hin + ((size_t)b * SEQ + (SEQ - TPF - 1) + j) * DM);
  ((uint2*)(outp + (size_t)row * DM))[threadIdx.x] = src[threadIdx.x];
}

__global__ __launch_bounds__(256) void death_kernel(
    const half_t* __restrict__ hin, const float* __restrict__ w,
    const float* __restrict__ bias, float* __restrict__ outp) {
  __shared__ float red[256];
  int b = blockIdx.x, t = threadIdx.x;
  const half_t* xr = hin + ((size_t)b * SEQ + SEQ - 1) * DM;
  float s = 0.0f;
  for (int i = t; i < DM; i += 256) s += (float)xr[i] * w[i];
  red[t] = s; __syncthreads();
  for (int o = 128; o > 0; o >>= 1) { if (t < o) red[t] += red[t + o]; __syncthreads(); }
  if (t == 0) outp[b] = red[0] + bias[0];
}

// ---------------- host orchestration ----------------
static inline void launch_cvt(const float* in, half_t* out, size_t n, hipStream_t s) {
  int n4 = (int)(n / 4);
  cvt_kernel<<<(n4 + 255) / 256, 256, 0, s>>>(in, out, n4);
}

extern "C" void kernel_launch(void* const* d_in, const int* in_sizes, int n_in,
                              void* d_out, int out_size, void* d_ws, size_t ws_size,
                              hipStream_t stream) {
  (void)in_sizes; (void)n_in; (void)out_size; (void)ws_size;
  const int*   ft       = (const int*)d_in[0];
  const int*   act      = (const int*)d_in[1];
  const int*   lvl      = (const int*)d_in[2];
  const float* tok_emb  = (const float*)d_in[3];
  const float* act_emb  = (const float*)d_in[4];
  const float* lvl_emb  = (const float*)d_in[5];
  const float* ln1_g    = (const float*)d_in[6];
  const float* ln1_b    = (const float*)d_in[7];
  const float* qkv_w    = (const float*)d_in[8];
  const float* qkv_b    = (const float*)d_in[9];
  const float* out_w    = (const float*)d_in[10];
  const float* out_b    = (const float*)d_in[11];
  const float* ln2_g    = (const float*)d_in[12];
  const float* ln2_b    = (const float*)d_in[13];
  const float* mlp_w1   = (const float*)d_in[14];
  const float* mlp_b1   = (const float*)d_in[15];
  const float* mlp_w2   = (const float*)d_in[16];
  const float* mlp_b2   = (const float*)d_in[17];
  const float* lnf_g    = (const float*)d_in[18];
  const float* lnf_b    = (const float*)d_in[19];
  const float* head_w   = (const float*)d_in[20];
  const float* death_w  = (const float*)d_in[21];
  const float* death_b  = (const float*)d_in[22];
  const float* rope_cos = (const float*)d_in[23];
  const float* rope_sin = (const float*)d_in[24];
  // d_in[25] = attn_mask: recomputed analytically in-kernel

  const int rows = BB * SEQ;                       // 11776 (= 92 * 128)
  // workspace partition (~254 MiB)
  char* ws = (char*)d_ws;
  float*  x     = (float*)ws;  ws += (size_t)rows * DM * 4;            // residual f32
  half_t* hb16  = (half_t*)ws; ws += (size_t)rows * DM * 2;            // LN out / attn out
  half_t* big16 = (half_t*)ws; ws += (size_t)rows * 4 * DM * 2;        // qkv / mlp hidden
  half_t* Qh    = (half_t*)ws; ws += (size_t)BB * NH * SEQ * HD * 2;
  half_t* Kh    = (half_t*)ws; ws += (size_t)BB * NH * SEQ * HD * 2;
  half_t* Vh    = (half_t*)ws; ws += (size_t)BB * NH * SEQ * HD * 2;
  half_t* wq16  = (half_t*)ws; ws += (size_t)3 * DM * DM * 2;          // per-layer f16 weights
  half_t* wo16  = (half_t*)ws; ws += (size_t)DM * DM * 2;
  half_t* w116  = (half_t*)ws; ws += (size_t)4 * DM * DM * 2;
  half_t* w216  = (half_t*)ws; ws += (size_t)4 * DM * DM * 2;
  float*  zbias = (float*)ws;  ws += (size_t)VOC * 4;
  float* logits = (float*)d_out;
  float* death  = logits + (size_t)BB * TPF * VOC;

  hipMemsetAsync(zbias, 0, (size_t)VOC * 4, stream);
  hipFuncSetAttribute((const void*)attn_kernel,
                      hipFuncAttributeMaxDynamicSharedMemorySize, ATTN_SMEM);
  hipFuncSetAttribute((const void*)gemm_f16,
                      hipFuncAttributeMaxDynamicSharedMemorySize, GSMEM);

  embed_kernel<<<rows, 256, 0, stream>>>(x, ft, act, lvl, tok_emb, act_emb, lvl_emb);

  for (int l = 0; l < NLAYER; l++) {
    launch_cvt(qkv_w  + (size_t)l * 3 * DM * DM, wq16, (size_t)3 * DM * DM, stream);
    launch_cvt(out_w  + (size_t)l * DM * DM,     wo16, (size_t)DM * DM,     stream);
    launch_cvt(mlp_w1 + (size_t)l * 4 * DM * DM, w116, (size_t)4 * DM * DM, stream);
    launch_cvt(mlp_w2 + (size_t)l * 4 * DM * DM, w216, (size_t)4 * DM * DM, stream);

    ln_kernel<<<rows, 256, 0, stream>>>(x, ln1_g + (size_t)l * DM, ln1_b + (size_t)l * DM, hb16);
    gemm_f16<<<dim3(rows / BM, (3 * DM) / BN), 256, GSMEM, stream>>>(
        hb16, wq16, qkv_b + (size_t)l * 3 * DM, big16, rows, 3 * DM, DM, FLAG_OUT16);
    rope_kernel<<<(BB * SEQ * NH * HD) / 256, 256, 0, stream>>>(
        big16, rope_cos, rope_sin, Qh, Kh, Vh);
    attn_kernel<<<BB * NH * 2, 192, ATTN_SMEM, stream>>>(Qh, Kh, Vh, hb16);
    gemm_f16<<<dim3(rows / BM, DM / BN), 256, GSMEM, stream>>>(
        hb16, wo16, out_b + (size_t)l * DM, x, rows, DM, DM, FLAG_RESID);
    ln_kernel<<<rows, 256, 0, stream>>>(x, ln2_g + (size_t)l * DM, ln2_b + (size_t)l * DM, hb16);
    gemm_f16<<<dim3(rows / BM, (4 * DM) / BN), 256, GSMEM, stream>>>(
        hb16, w116, mlp_b1 + (size_t)l * 4 * DM, big16, rows, 4 * DM, DM, FLAG_GELU | FLAG_OUT16);
    gemm_f16<<<dim3(rows / BM, DM / BN), 256, GSMEM, stream>>>(
        big16, w216, mlp_b2 + (size_t)l * DM, x, rows, DM, 4 * DM, FLAG_RESID);
  }

  ln_kernel<<<rows, 256, 0, stream>>>(x, lnf_g, lnf_b, hb16);
  gather_kernel<<<BB * TPF, 256, 0, stream>>>(hb16, big16);   // 2304 rows (= 18 * 128)
  launch_cvt(head_w, wq16, (size_t)VOC * DM, stream);         // reuse staging
  gemm_f16<<<dim3((BB * TPF) / BM, VOC / BN), 256, GSMEM, stream>>>(
      big16, wq16, zbias, logits, BB * TPF, VOC, DM, 0);
  death_kernel<<<BB, 256, 0, stream>>>(hb16, death_w, death_b, death);
}
